// AttentionPool_1288490189684
// MI455X (gfx1250) — matrix operand order repaired and checked by reading, not compiled, not run on hardware
//
#include <hip/hip_runtime.h>
#include <hip/hip_bf16.h>
#include <stdint.h>

#define N_NODES 262144
#define DIM     1024
#define HID     256
#define NSEG    2048

typedef __bf16 v16bf __attribute__((ext_vector_type(16)));
typedef float  v8f   __attribute__((ext_vector_type(8)));

// ---- helpers -------------------------------------------------------------

// fp32 -> bf16 RNE (scalar, used only in tiny pack kernel)
static __device__ __forceinline__ unsigned short f2bf(float f) {
    union { float f; unsigned u; } v; v.f = f;
    unsigned u = v.u;
    unsigned r = u + 0x7FFFu + ((u >> 16) & 1u);
    return (unsigned short)(r >> 16);
}

// packed fp32x2 -> bf16x2 (hot path: use v_cvt_pk_bf16_f32 when available)
#if __has_builtin(__builtin_amdgcn_cvt_pk_bf16_f32)
typedef __bf16 v2bf __attribute__((ext_vector_type(2)));
static __device__ __forceinline__ void cvt2(__bf16* d, float a, float b) {
    *(v2bf*)d = __builtin_amdgcn_cvt_pk_bf16_f32(a, b);
}
#else
static __device__ __forceinline__ void cvt2(__bf16* d, float a, float b) {
    d[0] = (__bf16)a; d[1] = (__bf16)b;
}
#endif

// monotonic float <-> uint key for atomicMax over signed floats
static __device__ __forceinline__ unsigned fkey(float f) {
    union { float f; unsigned u; } v; v.f = f;
    return (v.u & 0x80000000u) ? ~v.u : (v.u | 0x80000000u);
}
static __device__ __forceinline__ float funkey(unsigned k) {
    union { float f; unsigned u; } v;
    v.u = (k & 0x80000000u) ? (k ^ 0x80000000u) : ~k;
    return v.f;
}

// CDNA5 async global->LDS copy (ASYNCcnt-tracked, no VGPR round trip)
static __device__ __forceinline__ void async_b128(unsigned lds_off, const void* gaddr) {
    unsigned long long ga = (unsigned long long)(uintptr_t)gaddr;
    asm volatile("global_load_async_to_lds_b128 %0, %1, off"
                 :: "v"(lds_off), "v"(ga) : "memory");
}
static __device__ __forceinline__ void wait_async0() {
    asm volatile("s_wait_asynccnt 0" ::: "memory");
}

// ---- kernel 0: zero outputs / denom / max --------------------------------

__global__ void k_init(float* __restrict__ out, float* __restrict__ denom,
                       unsigned* __restrict__ smax) {
    int i = blockIdx.x * blockDim.x + threadIdx.x;
    if (i < NSEG * DIM) out[i] = 0.0f;
    if (i < NSEG)       denom[i] = 0.0f;
    if (i == 0)         *smax = 0u;   // below key of any finite float
}

// ---- kernel 1: pack W1 [DIM,HID] fp32 -> bf16 WMMA B-fragment layout -----
// Fragment order: [k_step(32)][n_tile(16)][lane(32)][elem(16)]
// B (32x16, 16-bit): N = lane & 15. elem idx e -> k_local = 8*(lane>>4)+(e&7)+16*(e>>3)
// (matches A: lanes 0-15 hold K=0..7,16..23; lanes 16-31 hold K=8..15,24..31).

__global__ void k_pack_w1(const float* __restrict__ W1,
                          unsigned short* __restrict__ w1p) {
    int i = blockIdx.x * blockDim.x + threadIdx.x;  // 0 .. DIM*HID-1
    int e    = i & 15;
    int lane = (i >> 4) & 31;
    int nt   = (i >> 9) & 15;
    int ks   = i >> 13;
    int n = nt * 16 + (lane & 15);
    int k = ks * 32 + ((lane >> 4) * 8) + (e & 7) + ((e >> 3) * 16);
    w1p[i] = f2bf(W1[k * HID + n]);
}

// ---- kernel 2: h = tanh(x@W1+b1); s = h@W2+b2; global max(s) -------------
// Block = 256 thr (8 waves), M_BLOCK = 128 rows, each wave owns 16 rows and
// 16 fp32 WMMA accumulators (full HID=256). B fragments staged into ping-pong
// LDS buffers with global_load_async_to_lds_b128; A (x) loads and B staging
// for step ks+1 overlap the 16-WMMA block of step ks.

__global__ __launch_bounds__(256, 1)
void k_mlp_score(const float* __restrict__ x,
                 const unsigned short* __restrict__ w1p,
                 const float* __restrict__ b1,
                 const float* __restrict__ W2,
                 const float* __restrict__ b2,
                 float* __restrict__ s,
                 unsigned* __restrict__ smax) {
    __shared__ uint4 ldsB[2048];  // 2 x 16 KB ping-pong

    const int tid  = threadIdx.x;
    const int lane = tid & 31;
    const int wave = tid >> 5;
    const int m    = lane & 15;   // A-row within tile / N within B tile
    const int hi   = lane >> 4;   // K-half selector
    const int rows0w = blockIdx.x * 128 + wave * 16;

    v8f acc[16];
#pragma unroll
    for (int nt = 0; nt < 16; ++nt)
        acc[nt] = (v8f){0.f, 0.f, 0.f, 0.f, 0.f, 0.f, 0.f, 0.f};

    const float* xrow = x + (size_t)(rows0w + m) * DIM;
    const int koff = hi * 8;

    // prologue: stage B(ks=0) into buffer 0; load A(ks=0)
#pragma unroll
    for (int j = 0; j < 4; ++j) {
        unsigned dst = (unsigned)(uintptr_t)&ldsB[tid + j * 256];
        async_b128(dst, (const uint4*)w1p + (tid + j * 256));
    }
    float4 ra0 = *(const float4*)(xrow + koff + 0);
    float4 ra1 = *(const float4*)(xrow + koff + 4);
    float4 ra2 = *(const float4*)(xrow + koff + 16);
    float4 ra3 = *(const float4*)(xrow + koff + 20);

    for (int ks = 0; ks < 32; ++ks) {
        wait_async0();        // our async loads for buffer ks are done
        __syncthreads();      // everyone's are; prev compute also done

        // stage B(ks+1) into the other buffer (overlaps this step's WMMAs)
        if (ks + 1 < 32) {
            const uint4* src = (const uint4*)w1p + (size_t)(ks + 1) * 1024;
            unsigned base = ((ks + 1) & 1) * 1024;
#pragma unroll
            for (int j = 0; j < 4; ++j) {
                unsigned dst = (unsigned)(uintptr_t)&ldsB[base + tid + j * 256];
                async_b128(dst, src + (tid + j * 256));
            }
        }

        // issue A(ks+1) global loads now, consumed next iteration
        const int ksn = (ks + 1 < 32) ? (ks + 1) : ks;
        const float* pn = xrow + ksn * 32 + koff;
        float4 na0 = *(const float4*)(pn + 0);
        float4 na1 = *(const float4*)(pn + 4);
        float4 na2 = *(const float4*)(pn + 16);
        float4 na3 = *(const float4*)(pn + 20);

        // convert current A tile to bf16 fragment
        union { __bf16 e[16]; v16bf v; } au;
        cvt2(au.e + 0,  ra0.x, ra0.y);
        cvt2(au.e + 2,  ra0.z, ra0.w);
        cvt2(au.e + 4,  ra1.x, ra1.y);
        cvt2(au.e + 6,  ra1.z, ra1.w);
        cvt2(au.e + 8,  ra2.x, ra2.y);
        cvt2(au.e + 10, ra2.z, ra2.w);
        cvt2(au.e + 12, ra3.x, ra3.y);
        cvt2(au.e + 14, ra3.z, ra3.w);

        // inner loop: double-buffered B-fragment regs so ds_load(nt+1)
        // overlaps wmma(nt)
        const uint4* bufc = ldsB + (ks & 1) * 1024;
        union Bu { uint4 q[2]; v16bf v; };
        Bu bq[2];
        bq[0].q[0] = bufc[(0 * 32 + lane) * 2 + 0];
        bq[0].q[1] = bufc[(0 * 32 + lane) * 2 + 1];
#pragma unroll
        for (int nt = 0; nt < 16; ++nt) {
            const int cb = nt & 1;
            if (nt < 15) {
                bq[cb ^ 1].q[0] = bufc[((nt + 1) * 32 + lane) * 2 + 0];
                bq[cb ^ 1].q[1] = bufc[((nt + 1) * 32 + lane) * 2 + 1];
            }
            acc[nt] = __builtin_amdgcn_wmma_f32_16x16x32_bf16(
                false, au.v, false, bq[cb].v, (short)0, acc[nt], false, false);
        }

        ra0 = na0; ra1 = na1; ra2 = na2; ra3 = na3;
    }

    // Epilogue: s[m'] = sum_n tanh(h[m',n] + b1[n]) * W2[n] + b2
    // C layout: VGPR r, lanes 0-15 -> M=r, N=lane; lanes 16-31 -> M=8+r.
    float b1v[16], w2v[16];
#pragma unroll
    for (int nt = 0; nt < 16; ++nt) {
        int n = nt * 16 + m;
        b1v[nt] = b1[n];
        w2v[nt] = W2[n];
    }
    const float bias2 = b2[0];
    float wmax = -3.4e38f;
#pragma unroll
    for (int r = 0; r < 8; ++r) {
        float p = 0.f;
#pragma unroll
        for (int nt = 0; nt < 16; ++nt)
            p += tanhf(acc[nt][r] + b1v[nt]) * w2v[nt];
        // reduce across the 16 lanes of each half (halves stay separate)
        p += __shfl_xor(p, 1, 32);
        p += __shfl_xor(p, 2, 32);
        p += __shfl_xor(p, 4, 32);
        p += __shfl_xor(p, 8, 32);
        if (m == 0) {
            float sv = p + bias2;
            s[rows0w + r + 8 * hi] = sv;
            wmax = fmaxf(wmax, sv);
        }
    }
    if (m == 0) atomicMax(smax, fkey(wmax));
}

// ---- kernel 3: exp_w = exp(s - max); denom = segment_sum(exp_w) ----------

__global__ void k_expdenom(const float* __restrict__ s,
                           const long long* __restrict__ batch,
                           const unsigned* __restrict__ smax,
                           float* __restrict__ expw,
                           float* __restrict__ denom) {
    int i = blockIdx.x * blockDim.x + threadIdx.x;
    if (i >= N_NODES) return;
    float mx = funkey(*smax);
    float w = expf(s[i] - mx);
    expw[i] = w;
    atomicAdd(&denom[(int)batch[i]], w);
}

// ---- kernel 4: out[seg] += attn * x  (sorted-segment register blocking) --

__global__ __launch_bounds__(256, 2)
void k_scatter(const float* __restrict__ x,
               const long long* __restrict__ batch,
               const float* __restrict__ expw,
               const float* __restrict__ denom,
               float* __restrict__ out) {
    const int tid  = threadIdx.x;
    const int row0 = blockIdx.x * 128;
    float4 acc = make_float4(0.f, 0.f, 0.f, 0.f);
    int cur = (int)batch[row0];
    for (int r = 0; r < 128; ++r) {
        int row = row0 + r;
        int seg = (int)batch[row];
        if (seg != cur) {
            float* o = out + (size_t)cur * DIM + tid * 4;
            atomicAdd(o + 0, acc.x); atomicAdd(o + 1, acc.y);
            atomicAdd(o + 2, acc.z); atomicAdd(o + 3, acc.w);
            acc = make_float4(0.f, 0.f, 0.f, 0.f);
            cur = seg;
        }
        float a = expw[row] / (denom[seg] + 1e-8f);
        float4 xv = ((const float4*)(x + (size_t)row * DIM))[tid];
        acc.x += a * xv.x; acc.y += a * xv.y;
        acc.z += a * xv.z; acc.w += a * xv.w;
    }
    float* o = out + (size_t)cur * DIM + tid * 4;
    atomicAdd(o + 0, acc.x); atomicAdd(o + 1, acc.y);
    atomicAdd(o + 2, acc.z); atomicAdd(o + 3, acc.w);
}

// ---- launcher ------------------------------------------------------------

extern "C" void kernel_launch(void* const* d_in, const int* in_sizes, int n_in,
                              void* d_out, int out_size, void* d_ws, size_t ws_size,
                              hipStream_t stream) {
    const float*     x     = (const float*)d_in[0];
    const long long* batch = (const long long*)d_in[1];
    const float*     W1    = (const float*)d_in[2];
    const float*     b1    = (const float*)d_in[3];
    const float*     W2    = (const float*)d_in[4];
    const float*     b2    = (const float*)d_in[5];
    float*           out   = (float*)d_out;

    // workspace layout (~2.6 MB)
    char* ws = (char*)d_ws;
    unsigned short* w1p   = (unsigned short*)(ws);                         // 512 KB
    float*          s     = (float*)(ws + (512 << 10));                    // 1 MB
    float*          expw  = (float*)(ws + (512 << 10) + (1 << 20));        // 1 MB
    float*          denom = (float*)(ws + (512 << 10) + (2 << 20));        // 8 KB
    unsigned*       smax  = (unsigned*)(ws + (512 << 10) + (2 << 20) + (8 << 10));

    k_init     <<<(NSEG * DIM + 255) / 256, 256, 0, stream>>>(out, denom, smax);
    k_pack_w1  <<<(DIM * HID) / 256,        256, 0, stream>>>(W1, w1p);
    k_mlp_score<<<N_NODES / 128,            256, 0, stream>>>(x, w1p, b1, W2, b2, s, smax);
    k_expdenom <<<N_NODES / 256,            256, 0, stream>>>(s, batch, smax, expw, denom);
    k_scatter  <<<N_NODES / 128,            256, 0, stream>>>(x, batch, expw, denom, out);
}
